// CSPNet_28286654612217
// MI455X (gfx1250) — matrix operand order repaired and checked by reading, not compiled, not run on hardware
//
#include <hip/hip_runtime.h>
#include <math.h>

#define G_GRAPHS 256
#define NPG 24
#define H 128
#define LAT 4
#define NLAYER 4
#define NFREQ 10
#define EDGE_IN 325
#define KP1 352                    // edge_w1 K padded (325 -> 352, mult of 32)
#define NN (G_GRAPHS*NPG)          // 6144 nodes
#define TWO_PI 6.283185307179586f

typedef __attribute__((ext_vector_type(16))) _Float16 v16h;
typedef __attribute__((ext_vector_type(8)))  float    v8f;

// ---------------- WMMA fragment helpers (wave32, 16x16x32 f16) ----------------
// A fragment: 16x32 (MxK) f16, row-major, leading dim `ld` halves.
// m = lane&15; lanes<16 hold K 0..15, lanes>=16 hold K 16..31.
__device__ __forceinline__ v16h load_a_h16(const _Float16* A, int ld) {
    int lane = threadIdx.x & 31;
    int m  = lane & 15;
    int kb = (lane & 16) ? 16 : 0;
    const _Float16* row = A + (size_t)m * ld + kb;
    v16h a;
#pragma unroll
    for (int i = 0; i < 8; ++i) a[i]     = row[i];      // 16B contiguous
#pragma unroll
    for (int i = 0; i < 8; ++i) a[8 + i] = row[8 + i];  // 16B contiguous
    return a;
}

// B fragment: 32x16 (KxN) f16 from PRE-TRANSPOSED half weights WT[N][Kld].
// base points at (column-block, k-block) origin: WT + ncol_base*Kld + k_base.
// n = lane&15; lanes<16 hold K 0..15, lanes>=16 hold K 16..31 -> 32B contiguous per lane.
__device__ __forceinline__ v16h load_b_t(const _Float16* base, int Kld) {
    int lane = threadIdx.x & 31;
    int n  = lane & 15;
    int kb = (lane & 16) ? 16 : 0;
    const _Float16* p = base + (size_t)n * Kld + kb;
    v16h b;
#pragma unroll
    for (int e = 0; e < 16; ++e) b[e] = p[e];
    return b;
}

// C/D fragment element (v in 0..7) -> (m,n).
__device__ __forceinline__ void dcoord(int v, int& m, int& n) {
    int lane = threadIdx.x & 31;
    n = lane & 15;
    m = (lane & 16) ? (8 + v) : v;
}

__device__ __forceinline__ float silu(float x) { return x / (1.0f + __expf(-x)); }

#define WMMA(a,b,c) __builtin_amdgcn_wmma_f32_16x16x32_f16(false, (a), false, (b), (short)0, (c), false, false)

// ---------------- prep: weights fp32 [L][K][H] -> half transposed [L][H][Kpad] ----------------
__global__ void k_wt(const float* __restrict__ W, int K, int Kpad, int Nc, int L,
                     _Float16* __restrict__ out) {
    int idx = blockIdx.x * blockDim.x + threadIdx.x;
    int per = Nc * Kpad;
    if (idx >= L * per) return;
    int l = idx / per, r = idx % per;
    int n = r / Kpad, k = r % Kpad;
    float v = (k < K) ? W[(size_t)l * K * Nc + (size_t)k * Nc + n] : 0.0f;
    out[idx] = (_Float16)v;
}

// ---------------- kernels ----------------

// h0 = [emb(atom), t(graph)] @ atom_latent_w + b   (small)
__global__ void k_init_h(const float* __restrict__ t, const int* __restrict__ atom_types,
                         const float* __restrict__ emb_table, const float* __restrict__ alw,
                         const float* __restrict__ alb,
                         float* __restrict__ h, _Float16* __restrict__ hh) {
    int node = blockIdx.x;
    int o = threadIdx.x;
    int g = node / NPG;
    int at = atom_types[node] - 1;
    const float* er = emb_table + (size_t)at * H;
    float acc = alb[o];
    for (int k = 0; k < H; ++k)   acc += er[k] * alw[(size_t)k * H + o];
    for (int k = 0; k < LAT; ++k) acc += t[g * LAT + k] * alw[(size_t)(H + k) * H + o];
    h[(size_t)node * H + o]  = acc;
    hh[(size_t)node * H + o] = (_Float16)acc;
}

// lat_ip[g, i*3+k] = <lf[g][i], lf[g][k]>
__global__ void k_latip(const float* __restrict__ lf, float* __restrict__ latip) {
    int idx = blockIdx.x * blockDim.x + threadIdx.x;
    if (idx >= G_GRAPHS * 9) return;
    int g = idx / 9, r = idx % 9, i = r / 3, k = r % 3;
    const float* a = lf + (size_t)g * 9;
    latip[idx] = a[i*3+0]*a[k*3+0] + a[i*3+1]*a[k*3+1] + a[i*3+2]*a[k*3+2];
}

// R[g][n] = edge_b1[l][n] + lat_ip[g] @ W1[256:265]
__global__ void k_R(const float* __restrict__ latip, const float* __restrict__ ew1,
                    const float* __restrict__ eb1, int l, float* __restrict__ R) {
    int idx = blockIdx.x * blockDim.x + threadIdx.x;   // G*H threads
    int g = idx / H, n = idx % H;
    const float* W = ew1 + (size_t)l * EDGE_IN * H;
    float s = eb1[l * H + n];
    for (int k = 0; k < 9; ++k) s += latip[g * 9 + k] * W[(size_t)(256 + k) * H + n];
    R[idx] = s;
}

// P = h @ W1[0:128], Q = h @ W1[128:256]   (WMMA; 64 nodes/block, 8 waves; wave owns column nt)
__global__ void k_node_pre(const _Float16* __restrict__ hh, const _Float16* __restrict__ ew1T,
                           int l, float* __restrict__ P, float* __restrict__ Q) {
    int nb = blockIdx.x * 64;
    int nt = threadIdx.x >> 5;   // wave id == output column tile
    const _Float16* WT = ew1T + (size_t)l * H * KP1 + (size_t)(nt * 16) * KP1;
#pragma unroll
    for (int pq = 0; pq < 2; ++pq) {
        v16h bf[4];
#pragma unroll
        for (int ks = 0; ks < 4; ++ks) bf[ks] = load_b_t(WT + pq * H + ks * 32, KP1);
        float* D = pq ? Q : P;
        for (int mt = 0; mt < 4; ++mt) {
            const _Float16* Arow = hh + (size_t)(nb + mt * 16) * H;
            v8f c = {};
#pragma unroll
            for (int ks = 0; ks < 4; ++ks) {
                v16h a = load_a_h16(Arow + ks * 32, H);
                c = WMMA(a, bf[ks], c);
            }
#pragma unroll
            for (int v = 0; v < 8; ++v) {
                int m, n; dcoord(v, m, n);
                D[(size_t)(nb + mt * 16 + m) * H + nt * 16 + n] = c[v];
            }
        }
    }
}

// Edge stage: one block per (graph, 4-src-node chunk) = 96 edges; wave owns column nt.
__global__ void k_edge(const float* __restrict__ frac,
                       const float* __restrict__ P, const float* __restrict__ Q,
                       const float* __restrict__ R,
                       const _Float16* __restrict__ ew1T, const _Float16* __restrict__ ew2T,
                       const float* __restrict__ eb2, int l, float* __restrict__ agg) {
    __shared__ float    sQ[NPG][H];     // 12 KB
    __shared__ float    sP[4][H];       //  2 KB
    __shared__ float    sR[H];
    __shared__ float    sF[NPG][3];
    __shared__ _Float16 sDis[96][64];   // 12 KB
    __shared__ _Float16 sEf[96][H];     // 24 KB
    __shared__ float    sAgg[4][H];     //  2 KB

    int g  = blockIdx.x / 6;
    int s0 = (blockIdx.x % 6) * 4;
    int gn = g * NPG;
    int tid = threadIdx.x;

    for (int i = tid; i < NPG * H; i += 256) sQ[i / H][i % H] = Q[(size_t)(gn + i / H) * H + i % H];
    for (int i = tid; i < 4 * H; i += 256) {
        sP[i / H][i % H] = P[(size_t)(gn + s0 + i / H) * H + i % H];
        sAgg[i / H][i % H] = 0.0f;
    }
    for (int i = tid; i < H; i += 256)        sR[i] = R[(size_t)g * H + i];
    for (int i = tid; i < NPG * 3; i += 256)  sF[i / 3][i % 3] = frac[(size_t)(gn + i / 3) * 3 + i % 3];
    __syncthreads();

    for (int i = tid; i < 96 * 64; i += 256) {
        int e = i >> 6, f = i & 63;
        float val = 0.0f;
        if (f < 60) {
            int fm = f % 30, c3 = fm / 10, fr = fm % 10;
            int il = s0 + e / 24, j = e % 24;
            float d = sF[j][c3] - sF[il][c3];
            float ang = TWO_PI * (float)fr * d;
            val = (f < 30) ? __sinf(ang) : __cosf(ang);
        }
        sDis[e][f] = (_Float16)val;
    }
    __syncthreads();

    int nt = tid >> 5;                       // wave id == column tile
    int lane_n = nt * 16 + (tid & 15);       // this lane's output column
    const _Float16* W1T = ew1T + (size_t)l * H * KP1 + (size_t)(nt * 16) * KP1;
    const _Float16* W2T = ew2T + (size_t)l * H * H  + (size_t)(nt * 16) * H;

    // GEMM1: [96,64] x [64,128], C init = P[src]+Q[dst]+R. B frags hoisted (rows 265..328, zero-padded).
    {
        v16h b0 = load_b_t(W1T + 265,      KP1);
        v16h b1 = load_b_t(W1T + 265 + 32, KP1);
        float rbias = sR[lane_n];
        for (int mt = 0; mt < 6; ++mt) {
            v8f c;
#pragma unroll
            for (int v = 0; v < 8; ++v) {
                int m, n; dcoord(v, m, n);
                int e = mt * 16 + m;
                c[v] = sP[e / 24][nt * 16 + n] + sQ[e % 24][nt * 16 + n] + rbias;
            }
            v16h a0 = load_a_h16(&sDis[mt * 16][0], 64);
            v16h a1 = load_a_h16(&sDis[mt * 16][0] + 32, 64);
            c = WMMA(a0, b0, c);
            c = WMMA(a1, b1, c);
#pragma unroll
            for (int v = 0; v < 8; ++v) {
                int m, n; dcoord(v, m, n);
                sEf[mt * 16 + m][nt * 16 + n] = (_Float16)silu(c[v]);
            }
        }
    }
    __syncthreads();

    // GEMM2: [96,128] x [128,128] + SiLU + LDS scatter-add by src. B frags hoisted.
    {
        v16h bf[4];
#pragma unroll
        for (int ks = 0; ks < 4; ++ks) bf[ks] = load_b_t(W2T + ks * 32, H);
        float bias = eb2[l * H + lane_n];
        for (int mt = 0; mt < 6; ++mt) {
            v8f c;
#pragma unroll
            for (int v = 0; v < 8; ++v) c[v] = bias;
#pragma unroll
            for (int ks = 0; ks < 4; ++ks) {
                v16h a = load_a_h16(&sEf[mt * 16][0] + ks * 32, H);
                c = WMMA(a, bf[ks], c);
            }
#pragma unroll
            for (int v = 0; v < 8; ++v) {
                int m, n; dcoord(v, m, n);
                int e = mt * 16 + m;
                atomicAdd(&sAgg[e / 24][nt * 16 + n], silu(c[v]));
            }
        }
    }
    __syncthreads();

    const float inv = 1.0f / (float)NPG;
    for (int i = tid; i < 4 * H; i += 256)
        agg[(size_t)(gn + s0 + i / H) * H + i % H] = sAgg[i / H][i % H] * inv;
}

// Node MLP: h += silu(silu([h,agg]@W1 + b1)@W2 + b2)   (WMMA; 64 nodes/block; wave owns column nt)
__global__ void k_node(const float* __restrict__ agg,
                       const _Float16* __restrict__ nw1T, const float* __restrict__ nb1,
                       const _Float16* __restrict__ nw2T, const float* __restrict__ nb2,
                       int l, float* __restrict__ h, _Float16* __restrict__ hh) {
    __shared__ _Float16 sNin[64][2 * H];   // 32 KB
    __shared__ _Float16 sUpd[64][H];       // 16 KB
    int nb = blockIdx.x * 64;
    int tid = threadIdx.x;
    for (int i = tid; i < 64 * 2 * H; i += 256) {
        int r = i / (2 * H), cc = i % (2 * H);
        float v = (cc < H) ? h[(size_t)(nb + r) * H + cc]
                           : agg[(size_t)(nb + r) * H + (cc - H)];
        sNin[r][cc] = (_Float16)v;
    }
    __syncthreads();

    int nt = tid >> 5;
    int lane_n = nt * 16 + (tid & 15);
    const _Float16* W1T = nw1T + (size_t)l * H * (2 * H) + (size_t)(nt * 16) * (2 * H);
    const _Float16* W2T = nw2T + (size_t)l * H * H       + (size_t)(nt * 16) * H;

    {
        v16h bf[8];
#pragma unroll
        for (int ks = 0; ks < 8; ++ks) bf[ks] = load_b_t(W1T + ks * 32, 2 * H);
        float bias = nb1[l * H + lane_n];
        for (int mt = 0; mt < 4; ++mt) {
            v8f c;
#pragma unroll
            for (int v = 0; v < 8; ++v) c[v] = bias;
#pragma unroll
            for (int ks = 0; ks < 8; ++ks) {
                v16h a = load_a_h16(&sNin[mt * 16][0] + ks * 32, 2 * H);
                c = WMMA(a, bf[ks], c);
            }
#pragma unroll
            for (int v = 0; v < 8; ++v) {
                int m, n; dcoord(v, m, n);
                sUpd[mt * 16 + m][nt * 16 + n] = (_Float16)silu(c[v]);
            }
        }
    }
    __syncthreads();

    {
        v16h bf[4];
#pragma unroll
        for (int ks = 0; ks < 4; ++ks) bf[ks] = load_b_t(W2T + ks * 32, H);
        float bias = nb2[l * H + lane_n];
        for (int mt = 0; mt < 4; ++mt) {
            v8f c;
#pragma unroll
            for (int v = 0; v < 8; ++v) c[v] = bias;
#pragma unroll
            for (int ks = 0; ks < 4; ++ks) {
                v16h a = load_a_h16(&sUpd[mt * 16][0] + ks * 32, H);
                c = WMMA(a, bf[ks], c);
            }
#pragma unroll
            for (int v = 0; v < 8; ++v) {
                int m, n; dcoord(v, m, n);
                size_t idx = (size_t)(nb + mt * 16 + m) * H + nt * 16 + n;
                float hv = h[idx] + silu(c[v]);
                h[idx]  = hv;
                hh[idx] = (_Float16)hv;
            }
        }
    }
}

__global__ void k_coord(const float* __restrict__ h, const float* __restrict__ cw,
                        float* __restrict__ out) {
    int idx = blockIdx.x * blockDim.x + threadIdx.x;
    if (idx >= NN * 3) return;
    int node = idx / 3, d = idx % 3;
    float s = 0.0f;
    for (int k = 0; k < H; ++k) s += h[(size_t)node * H + k] * cw[k * 3 + d];
    out[G_GRAPHS * 9 + idx] = s;
}

__global__ void k_lattice(const float* __restrict__ h, const float* __restrict__ lw,
                          const float* __restrict__ lattices, float* __restrict__ out) {
    __shared__ float gf[H];
    __shared__ float lo[9];
    int g = blockIdx.x, n = threadIdx.x;
    float s = 0.0f;
    for (int a = 0; a < NPG; ++a) s += h[(size_t)(g * NPG + a) * H + n];
    gf[n] = s / (float)NPG;
    __syncthreads();
    if (n < 9) {
        float t = 0.0f;
        for (int k = 0; k < H; ++k) t += gf[k] * lw[k * 9 + n];
        lo[n] = t;
    }
    __syncthreads();
    if (n < 9) {
        int i = n / 3, kc = n % 3;
        out[g * 9 + n] = lo[i * 3 + 0] * lattices[(size_t)g * 9 + 0 * 3 + kc]
                       + lo[i * 3 + 1] * lattices[(size_t)g * 9 + 1 * 3 + kc]
                       + lo[i * 3 + 2] * lattices[(size_t)g * 9 + 2 * 3 + kc];
    }
}

extern "C" void kernel_launch(void* const* d_in, const int* in_sizes, int n_in,
                              void* d_out, int out_size, void* d_ws, size_t ws_size,
                              hipStream_t stream) {
    (void)in_sizes; (void)n_in; (void)out_size; (void)ws_size;
    const float* t         = (const float*)d_in[0];
    const float* frac      = (const float*)d_in[1];
    const float* lat_feats = (const float*)d_in[2];
    const float* lattices  = (const float*)d_in[3];
    const int*   atypes    = (const int*)d_in[4];
    // d_in[5]=node2graph, d_in[6]=edge_index: fc structure is analytic (edge = i*NPG+j), unused
    const float* emb_table = (const float*)d_in[7];
    const float* alw       = (const float*)d_in[8];
    const float* alb       = (const float*)d_in[9];
    const float* ew1       = (const float*)d_in[10];
    const float* eb1       = (const float*)d_in[11];
    const float* ew2       = (const float*)d_in[12];
    const float* eb2       = (const float*)d_in[13];
    const float* nw1       = (const float*)d_in[14];
    const float* nb1       = (const float*)d_in[15];
    const float* nw2       = (const float*)d_in[16];
    const float* nb2       = (const float*)d_in[17];
    const float* cw        = (const float*)d_in[18];
    const float* lw        = (const float*)d_in[19];
    float* out = (float*)d_out;

    char* ws = (char*)d_ws;
    size_t off = 0;
    float*    h    = (float*)(ws + off);    off += (size_t)NN * H * 4;
    _Float16* hh   = (_Float16*)(ws + off); off += (size_t)NN * H * 2;
    float*    P    = (float*)(ws + off);    off += (size_t)NN * H * 4;
    float*    Q    = (float*)(ws + off);    off += (size_t)NN * H * 4;
    float*    agg  = (float*)(ws + off);    off += (size_t)NN * H * 4;
    float*    R    = (float*)(ws + off);    off += (size_t)G_GRAPHS * H * 4;
    float*    lip  = (float*)(ws + off);    off += (size_t)G_GRAPHS * 9 * 4;
    _Float16* ew1T = (_Float16*)(ws + off); off += (size_t)NLAYER * H * KP1 * 2;
    _Float16* ew2T = (_Float16*)(ws + off); off += (size_t)NLAYER * H * H * 2;
    _Float16* nw1T = (_Float16*)(ws + off); off += (size_t)NLAYER * H * 2 * H * 2;
    _Float16* nw2T = (_Float16*)(ws + off); off += (size_t)NLAYER * H * H * 2;

    // weight prep (transpose + fp32->f16), once per launch
    {
        int n1 = NLAYER * H * KP1;
        k_wt<<<(n1 + 255) / 256, 256, 0, stream>>>(ew1, EDGE_IN, KP1, H, NLAYER, ew1T);
        int n2 = NLAYER * H * H;
        k_wt<<<(n2 + 255) / 256, 256, 0, stream>>>(ew2, H, H, H, NLAYER, ew2T);
        int n3 = NLAYER * H * 2 * H;
        k_wt<<<(n3 + 255) / 256, 256, 0, stream>>>(nw1, 2 * H, 2 * H, H, NLAYER, nw1T);
        k_wt<<<(n2 + 255) / 256, 256, 0, stream>>>(nw2, H, H, H, NLAYER, nw2T);
    }

    k_init_h<<<NN, H, 0, stream>>>(t, atypes, emb_table, alw, alb, h, hh);
    k_latip<<<(G_GRAPHS * 9 + 255) / 256, 256, 0, stream>>>(lat_feats, lip);
    for (int l = 0; l < NLAYER; ++l) {
        k_node_pre<<<NN / 64, 256, 0, stream>>>(hh, ew1T, l, P, Q);
        k_R<<<(G_GRAPHS * H) / 256, 256, 0, stream>>>(lip, ew1, eb1, l, R);
        k_edge<<<G_GRAPHS * 6, 256, 0, stream>>>(frac, P, Q, R, ew1T, ew2T, eb2, l, agg);
        k_node<<<NN / 64, 256, 0, stream>>>(agg, nw1T, nb1, nw2T, nb2, l, h, hh);
    }
    k_coord<<<(NN * 3 + 255) / 256, 256, 0, stream>>>(h, cw, out);
    k_lattice<<<G_GRAPHS, H, 0, stream>>>(h, lw, lattices, out);
}
